// WeightedAttention_67233418051998
// MI455X (gfx1250) — matrix operand order repaired and checked by reading, not compiled, not run on hardware
//
#include <hip/hip_runtime.h>
#include <hip/hip_bf16.h>
#include <cmath>

typedef __bf16 v16bf __attribute__((ext_vector_type(16)));
typedef __bf16 v8bf  __attribute__((ext_vector_type(8)));
typedef float  v8f   __attribute__((ext_vector_type(8)));

#define EPS_ 1e-8f
#define LN_EPS_ 1e-5f

// -------------------------------------------------------------------------
// WMMA helpers (CDNA5 wave32, bf16 16x16x32, f32 accumulate)
// -------------------------------------------------------------------------
__device__ __forceinline__ v8f wmma_bf16f32(v16bf a, v16bf b, v8f c) {
  // 8 args: (neg_a, A, neg_b, B, c_mod, C, reuse_a, reuse_b)
  return __builtin_amdgcn_wmma_f32_16x16x32_bf16(false, a, false, b, (short)0, c, false, false);
}

// A fragment: 16x32 bf16 tile from LDS, row-major [row][k], stride 32.
// ISA layout: lanes 0-15 hold row=lane, K {0..7,16..23}; lanes 16-31 row=lane-16, K {8..15,24..31}.
__device__ __forceinline__ v16bf load_a_frag(const __bf16* lds, int lane, int row0) {
  const int row = row0 + (lane & 15);
  const int kb  = (lane >> 4) * 8;
  v16bf a;
  ((v8bf*)&a)[0] = *(const v8bf*)(lds + row * 32 + kb);
  ((v8bf*)&a)[1] = *(const v8bf*)(lds + row * 32 + kb + 16);
  return a;
}

// B fragment: 32x16 bf16 tile, from LDS holding B-transposed rows [n][k], stride 32.
// Lane holds column n=lane&15, K {0..7,16..23} (lanes 0-15) / {8..15,24..31} (lanes 16-31).
__device__ __forceinline__ v16bf load_b_frag(const __bf16* lds, int lane, int col0) {
  const int n  = col0 + (lane & 15);
  const int kb = (lane >> 4) * 8;
  v16bf b;
  ((v8bf*)&b)[0] = *(const v8bf*)(lds + n * 32 + kb);
  ((v8bf*)&b)[1] = *(const v8bf*)(lds + n * 32 + kb + 16);
  return b;
}

// -------------------------------------------------------------------------
// CDNA5 async global->LDS DMA (tracked by ASYNCcnt). Per-lane 16B copy.
// -------------------------------------------------------------------------
__device__ __forceinline__ void async_copy_b128(unsigned lds_off, const void* gaddr) {
  asm volatile("global_load_async_to_lds_b128 %0, %1, off"
               :: "v"(lds_off), "v"(gaddr) : "memory");
}
__device__ __forceinline__ void wait_async0() {
  asm volatile("s_wait_asynccnt 0" ::: "memory");
}
// Allow the most recent 6 async ops (one tile-stage group) to stay in flight:
// waiting <=6 guarantees the PREVIOUS stage group has landed in LDS.
__device__ __forceinline__ void wait_async_prev6() {
  asm volatile("s_wait_asynccnt 6" ::: "memory");
}

// Stage a ROWS x 32 bf16 tile (global row stride LDK elems) into LDS via async DMA.
// 128-thread cooperative: ROWS*4 16B chunks. (ROWS=64 -> 2 ops/thr, 128 -> 4.)
template<int ROWS, int LDK>
__device__ __forceinline__ void stage_tile_async(const __bf16* g, __bf16* lds, int t) {
  #pragma unroll
  for (int cc = 0; cc < (ROWS * 4) / 128; ++cc) {
    const int idx = t + cc * 128;
    const int row = idx >> 2, ch = (idx & 3) * 8;
    async_copy_b128((unsigned)(size_t)(lds + row * 32 + ch),
                    (const void*)(g + (size_t)row * LDK + ch));
  }
}

// -------------------------------------------------------------------------
// Kernel 1: LayerNorm rows of 512 f32 -> bf16.  One block (128 thr) per row.
// -------------------------------------------------------------------------
__global__ void wa_layernorm_bf16(const float* __restrict__ x,
                                  const float* __restrict__ g,
                                  const float* __restrict__ bta,
                                  __bf16* __restrict__ out) {
  const size_t row = blockIdx.x;
  const int t = threadIdx.x;               // 0..127, 4 floats each = 512
  const float4 v = ((const float4*)(x + row * 512))[t];
  float s  = v.x + v.y + v.z + v.w;
  float sq = v.x * v.x + v.y * v.y + v.z * v.z + v.w * v.w;
  #pragma unroll
  for (int o = 16; o > 0; o >>= 1) { s += __shfl_down(s, o); sq += __shfl_down(sq, o); }
  __shared__ float ss[4], ssq[4];
  const int w = t >> 5, ln = t & 31;
  if (ln == 0) { ss[w] = s; ssq[w] = sq; }
  __syncthreads();
  if (t == 0) {
    ss[0]  = ss[0] + ss[1] + ss[2] + ss[3];
    ssq[0] = ssq[0] + ssq[1] + ssq[2] + ssq[3];
  }
  __syncthreads();
  const float mu   = ss[0] * (1.0f / 512.0f);
  const float var  = ssq[0] * (1.0f / 512.0f) - mu * mu;
  const float rstd = rsqrtf(var + LN_EPS_);
  const float4 gv = ((const float4*)g)[t];
  const float4 bv = ((const float4*)bta)[t];
  __bf16* op = out + row * 512 + t * 4;
  op[0] = (__bf16)((v.x - mu) * rstd * gv.x + bv.x);
  op[1] = (__bf16)((v.y - mu) * rstd * gv.y + bv.y);
  op[2] = (__bf16)((v.z - mu) * rstd * gv.z + bv.z);
  op[3] = (__bf16)((v.w - mu) * rstd * gv.w + bv.w);
}

// -------------------------------------------------------------------------
// Kernel 2: 512x512 weight transpose + f32->bf16.
// -------------------------------------------------------------------------
__global__ void wa_transpose_w_bf16(const float* __restrict__ W, __bf16* __restrict__ Wt) {
  __shared__ float tile[32][33];
  const int tx = threadIdx.x, ty = threadIdx.y;
  const int n0 = blockIdx.x * 32, k0 = blockIdx.y * 32;
  tile[ty][tx] = W[(size_t)(k0 + ty) * 512 + n0 + tx];
  __syncthreads();
  Wt[(size_t)(n0 + ty) * 512 + k0 + tx] = (__bf16)tile[tx][ty];
}

// -------------------------------------------------------------------------
// Kernel 3: C[M,Nc](bf16) = A[M,Kc](bf16) @ Bt[Nc,Kc]^T + bias.
// 64x128 block tile, 4 waves, each wave 32x64 via 2x4 WMMA frags.
// Double-buffered LDS: stage k+1 via async DMA while computing k
// (s_wait_asynccnt 6 keeps the in-flight group outstanding).
// -------------------------------------------------------------------------
template<int Kc, int Nc>
__global__ void wa_gemm_proj(const __bf16* __restrict__ A,
                             const __bf16* __restrict__ Bt,
                             const float* __restrict__ bias,
                             __bf16* __restrict__ C) {
  __shared__ __bf16 As[2][64 * 32];
  __shared__ __bf16 Bs[2][128 * 32];
  const int t = threadIdx.x, lane = t & 31, w = t >> 5;
  const int wm = (w >> 1) * 32, wn = (w & 1) * 64;
  const int m0 = blockIdx.y * 64, n0 = blockIdx.x * 128;
  const __bf16* Ab = A  + (size_t)m0 * Kc;
  const __bf16* Bb = Bt + (size_t)n0 * Kc;
  v8f acc[2][4] = {};
  // prologue: stage k0 = 0 into buffer 0
  stage_tile_async<64,  Kc>(Ab, As[0], t);
  stage_tile_async<128, Kc>(Bb, Bs[0], t);
  for (int k0 = 0; k0 < Kc; k0 += 32) {
    const int cur = (k0 >> 5) & 1, nxt = cur ^ 1;
    if (k0 + 32 < Kc) {
      stage_tile_async<64,  Kc>(Ab + k0 + 32, As[nxt], t);
      stage_tile_async<128, Kc>(Bb + k0 + 32, Bs[nxt], t);
      wait_async_prev6();      // current buffer's group has landed
    } else {
      wait_async0();
    }
    __syncthreads();
    v16bf af[2], bf[4];
    #pragma unroll
    for (int mi = 0; mi < 2; ++mi) af[mi] = load_a_frag(As[cur], lane, wm + mi * 16);
    #pragma unroll
    for (int ni = 0; ni < 4; ++ni) bf[ni] = load_b_frag(Bs[cur], lane, wn + ni * 16);
    #pragma unroll
    for (int mi = 0; mi < 2; ++mi)
      #pragma unroll
      for (int ni = 0; ni < 4; ++ni)
        acc[mi][ni] = wmma_bf16f32(af[mi], bf[ni], acc[mi][ni]);
    __syncthreads();
  }
  #pragma unroll
  for (int mi = 0; mi < 2; ++mi)
    #pragma unroll
    for (int ni = 0; ni < 4; ++ni) {
      const int n  = n0 + wn + ni * 16 + (lane & 15);
      const int mb = m0 + wm + mi * 16 + (lane >> 4) * 8;
      const float bv = bias[n];
      __bf16* cp = C + (size_t)mb * Nc + n;
      #pragma unroll
      for (int r = 0; r < 8; ++r) cp[r * Nc] = (__bf16)(acc[mi][ni][r] + bv);
    }
}

// -------------------------------------------------------------------------
// Kernel 4: dotsT[b,j,i] = scale * (k[b,j,:].q[b,i,:]), f32 out, batched z.
// Same double-buffered async pipeline.
// -------------------------------------------------------------------------
__global__ void wa_gemm_scores(const __bf16* __restrict__ Kh,
                               const __bf16* __restrict__ Qh,
                               float* __restrict__ dT, float scale) {
  __shared__ __bf16 As[2][64 * 32];
  __shared__ __bf16 Bs[2][128 * 32];
  const int t = threadIdx.x, lane = t & 31, w = t >> 5;
  const int wm = (w >> 1) * 32, wn = (w & 1) * 64;
  const int m0 = blockIdx.y * 64, n0 = blockIdx.x * 128;
  const __bf16* Ab = Kh + (size_t)blockIdx.z * 2048 * 512 + (size_t)m0 * 512;
  const __bf16* Bb = Qh + (size_t)blockIdx.z * 2048 * 512 + (size_t)n0 * 512;
  float* C = dT + (size_t)blockIdx.z * 2048 * 2048;
  v8f acc[2][4] = {};
  stage_tile_async<64,  512>(Ab, As[0], t);
  stage_tile_async<128, 512>(Bb, Bs[0], t);
  for (int k0 = 0; k0 < 512; k0 += 32) {
    const int cur = (k0 >> 5) & 1, nxt = cur ^ 1;
    if (k0 + 32 < 512) {
      stage_tile_async<64,  512>(Ab + k0 + 32, As[nxt], t);
      stage_tile_async<128, 512>(Bb + k0 + 32, Bs[nxt], t);
      wait_async_prev6();
    } else {
      wait_async0();
    }
    __syncthreads();
    v16bf af[2], bf[4];
    #pragma unroll
    for (int mi = 0; mi < 2; ++mi) af[mi] = load_a_frag(As[cur], lane, wm + mi * 16);
    #pragma unroll
    for (int ni = 0; ni < 4; ++ni) bf[ni] = load_b_frag(Bs[cur], lane, wn + ni * 16);
    #pragma unroll
    for (int mi = 0; mi < 2; ++mi)
      #pragma unroll
      for (int ni = 0; ni < 4; ++ni)
        acc[mi][ni] = wmma_bf16f32(af[mi], bf[ni], acc[mi][ni]);
    __syncthreads();
  }
  #pragma unroll
  for (int mi = 0; mi < 2; ++mi)
    #pragma unroll
    for (int ni = 0; ni < 4; ++ni) {
      const int n  = n0 + wn + ni * 16 + (lane & 15);
      const int mb = m0 + wm + mi * 16 + (lane >> 4) * 8;
      float* cp = C + (size_t)mb * 2048 + n;
      #pragma unroll
      for (int r = 0; r < 8; ++r) cp[r * 2048] = acc[mi][ni][r] * scale;
    }
}

// -------------------------------------------------------------------------
// Kernel 5: per (b,j) softmax stats over i (contiguous rows of dotsT).
// -------------------------------------------------------------------------
__global__ void wa_col_softmax_stats(const float* __restrict__ dT,
                                     float* __restrict__ mx,
                                     float* __restrict__ seinv) {
  __shared__ float red[8];
  const size_t row = blockIdx.x + (size_t)blockIdx.y * 2048;   // b*2048 + j
  const float* p = dT + row * 2048;
  const int t = threadIdx.x;
  float vals[8];
  float m = -INFINITY;
  #pragma unroll
  for (int k = 0; k < 8; ++k) { vals[k] = p[t + k * 256]; m = fmaxf(m, vals[k]); }
  #pragma unroll
  for (int o = 16; o > 0; o >>= 1) m = fmaxf(m, __shfl_down(m, o));
  if ((t & 31) == 0) red[t >> 5] = m;
  __syncthreads();
  if (t < 32) {
    float x = (t < 8) ? red[t] : -INFINITY;
    #pragma unroll
    for (int o = 4; o > 0; o >>= 1) x = fmaxf(x, __shfl_down(x, o));
    if (t == 0) red[0] = x;
  }
  __syncthreads();
  m = red[0];
  __syncthreads();
  float s = 0.0f;
  #pragma unroll
  for (int k = 0; k < 8; ++k) s += __expf(vals[k] - m);
  #pragma unroll
  for (int o = 16; o > 0; o >>= 1) s += __shfl_down(s, o);
  if ((t & 31) == 0) red[t >> 5] = s;
  __syncthreads();
  if (t < 32) {
    float x = (t < 8) ? red[t] : 0.0f;
    #pragma unroll
    for (int o = 4; o > 0; o >>= 1) x += __shfl_down(x, o);
    if (t == 0) { mx[row] = m; seinv[row] = 1.0f / x; }
  }
}

__global__ void wa_zero_f32(float* __restrict__ p, int n) {
  const int i = blockIdx.x * blockDim.x + threadIdx.x;
  if (i < n) p[i] = 0.0f;
}

// -------------------------------------------------------------------------
// Kernel 6: rs[b,i] = sum_j (exp(dT[j,i]-mx[j])*seinv[j] + EPS).
// Tile [64 j][256 i], coalesced over i, atomic accumulate.
// -------------------------------------------------------------------------
__global__ void wa_row_sums(const float* __restrict__ dT,
                            const float* __restrict__ mx,
                            const float* __restrict__ seinv,
                            float* __restrict__ rs) {
  __shared__ float smx[64], sse[64];
  const int t = threadIdx.x;
  const int i0 = blockIdx.x * 256, j0 = blockIdx.y * 64, b = blockIdx.z;
  const size_t base = (size_t)b * 2048 * 2048;
  if (t < 64) { smx[t] = mx[b * 2048 + j0 + t]; sse[t] = seinv[b * 2048 + j0 + t]; }
  __syncthreads();
  const int i = i0 + t;
  float acc = 0.0f;
  #pragma unroll 4
  for (int j = 0; j < 64; ++j) {
    const float v = dT[base + (size_t)(j0 + j) * 2048 + i];
    acc += __expf(v - smx[j]) * sse[j] + EPS_;
  }
  atomicAdd(&rs[b * 2048 + i], acc);
}

// -------------------------------------------------------------------------
// Kernel 7: updates[b,i,d] = sum_j mean_attn[i,j] * v[b,j,d].
// attn recomputed on the fly; A tile [64 i][32 j] bf16 built in LDS,
// v^T tile [128 d][32 j] bf16 staged; 64x128 block, 2x4 WMMA per wave.
// -------------------------------------------------------------------------
__global__ void wa_attn_v_gemm(const float* __restrict__ dT,
                               const float* __restrict__ mx,
                               const float* __restrict__ seinv,
                               const float* __restrict__ rs,
                               const __bf16* __restrict__ Vh,
                               float* __restrict__ out) {
  __shared__ __bf16 As[64 * 32];    // attn tile [i][j]
  __shared__ __bf16 Bs[128 * 32];   // v^T tile  [d][j]
  __shared__ float rinv[64];
  const int t = threadIdx.x, lane = t & 31, w = t >> 5;
  const int wm = (w >> 1) * 32, wn = (w & 1) * 64;
  const int m0 = blockIdx.y * 64;          // i tile
  const int n0 = blockIdx.x * 128;         // d tile
  const int b = blockIdx.z;
  const size_t dbase = (size_t)b * 2048 * 2048;
  const size_t vbase = (size_t)b * 2048 * 512;
  if (t < 64) rinv[t] = 1.0f / (rs[b * 2048 + m0 + t] + EPS_);
  __syncthreads();
  v8f acc[2][4] = {};
  for (int j0 = 0; j0 < 2048; j0 += 32) {
    // build attn A-tile: read dT rows (contiguous over i), transpose into LDS
    #pragma unroll
    for (int r = 0; r < 16; ++r) {
      const int lin = r * 128 + t;
      const int j = lin >> 6, ii = lin & 63;
      const float v = dT[dbase + (size_t)(j0 + j) * 2048 + m0 + ii];
      const float mj = mx[b * 2048 + j0 + j];
      const float sj = seinv[b * 2048 + j0 + j];
      As[ii * 32 + j] = (__bf16)((__expf(v - mj) * sj + EPS_) * rinv[ii]);
    }
    // build v^T B-tile (128 d x 32 j), coalesced reads over d
    #pragma unroll
    for (int r = 0; r < 32; ++r) {
      const int lin = r * 128 + t;
      const int j = lin >> 7, dd = lin & 127;
      Bs[dd * 32 + j] = Vh[vbase + (size_t)(j0 + j) * 512 + n0 + dd];
    }
    __syncthreads();
    v16bf af[2], bf[4];
    #pragma unroll
    for (int mi = 0; mi < 2; ++mi) af[mi] = load_a_frag(As, lane, wm + mi * 16);
    #pragma unroll
    for (int ni = 0; ni < 4; ++ni) bf[ni] = load_b_frag(Bs, lane, wn + ni * 16);
    #pragma unroll
    for (int mi = 0; mi < 2; ++mi)
      #pragma unroll
      for (int ni = 0; ni < 4; ++ni)
        acc[mi][ni] = wmma_bf16f32(af[mi], bf[ni], acc[mi][ni]);
    __syncthreads();
  }
  #pragma unroll
  for (int mi = 0; mi < 2; ++mi)
    #pragma unroll
    for (int ni = 0; ni < 4; ++ni) {
      const int n  = n0 + wn + ni * 16 + (lane & 15);
      const int mb = m0 + wm + mi * 16 + (lane >> 4) * 8;
      float* cp = out + (size_t)b * 2048 * 512 + (size_t)mb * 512 + n;
      #pragma unroll
      for (int r = 0; r < 8; ++r) cp[r * 512] = acc[mi][ni][r];
    }
}

// -------------------------------------------------------------------------
// Host launcher
// -------------------------------------------------------------------------
extern "C" void kernel_launch(void* const* d_in, const int* in_sizes, int n_in,
                              void* d_out, int out_size, void* d_ws, size_t ws_size,
                              hipStream_t stream) {
  const float* inputs  = (const float*)d_in[0];
  const float* context = (const float*)d_in[1];
  const float* g_in    = (const float*)d_in[2];
  const float* b_in    = (const float*)d_in[3];
  const float* g_ctx   = (const float*)d_in[4];
  const float* b_ctx   = (const float*)d_in[5];
  const float* Wq      = (const float*)d_in[6];
  const float* bq      = (const float*)d_in[7];
  const float* Wk      = (const float*)d_in[8];
  const float* bk      = (const float*)d_in[9];
  const float* Wv      = (const float*)d_in[10];
  const float* bv      = (const float*)d_in[11];
  float* out = (float*)d_out;

  const int B = 8, I = 2048, J = 2048, D = 512;
  const int R = B * I;                 // 16384 rows
  const float scale = 0.044194173824159216f;  // 512^-0.5

  // workspace carve-up (256B aligned)
  char* base = (char*)d_ws;
  size_t off = 0;
  auto carve = [&](size_t bytes) -> char* {
    char* p = base + off;
    off = (off + bytes + 255) & ~(size_t)255;
    return p;
  };
  __bf16* xh  = (__bf16*)carve((size_t)R * D * 2);
  __bf16* ch  = (__bf16*)carve((size_t)R * D * 2);
  __bf16* wqt = (__bf16*)carve((size_t)D * D * 2);
  __bf16* wkt = (__bf16*)carve((size_t)D * D * 2);
  __bf16* wvt = (__bf16*)carve((size_t)D * D * 2);
  __bf16* qh  = (__bf16*)carve((size_t)R * D * 2);
  __bf16* kh  = (__bf16*)carve((size_t)R * D * 2);
  __bf16* vh  = (__bf16*)carve((size_t)R * D * 2);
  float*  dT  = (float*)carve((size_t)B * J * I * 4);
  float*  mx  = (float*)carve((size_t)B * J * 4);
  float*  sei = (float*)carve((size_t)B * J * 4);
  float*  rs  = (float*)carve((size_t)B * I * 4);

  // 1) layernorms -> bf16
  wa_layernorm_bf16<<<R, 128, 0, stream>>>(inputs, g_in, b_in, xh);
  wa_layernorm_bf16<<<R, 128, 0, stream>>>(context, g_ctx, b_ctx, ch);

  // 2) weight transposes -> bf16
  dim3 tb(32, 32), tg(16, 16);
  wa_transpose_w_bf16<<<tg, tb, 0, stream>>>(Wq, wqt);
  wa_transpose_w_bf16<<<tg, tb, 0, stream>>>(Wk, wkt);
  wa_transpose_w_bf16<<<tg, tb, 0, stream>>>(Wv, wvt);

  // 3) q/k/v projections (WMMA, double-buffered async DMA)
  dim3 gQKV(D / 128, R / 64);
  wa_gemm_proj<512, 512><<<gQKV, 128, 0, stream>>>(xh, wqt, bq, qh);
  wa_gemm_proj<512, 512><<<gQKV, 128, 0, stream>>>(ch, wkt, bk, kh);
  wa_gemm_proj<512, 512><<<gQKV, 128, 0, stream>>>(ch, wvt, bv, vh);

  // 4) dotsT = scale * k @ q^T  (WMMA, batched, double-buffered)
  wa_gemm_scores<<<dim3(I / 128, J / 64, B), 128, 0, stream>>>(kh, qh, dT, scale);

  // 5) softmax stats over query axis (rows of dotsT)
  wa_col_softmax_stats<<<dim3(J, B), 256, 0, stream>>>(dT, mx, sei);

  // 6) per-query row sums of attn
  wa_zero_f32<<<(B * I + 255) / 256, 256, 0, stream>>>(rs, B * I);
  wa_row_sums<<<dim3(I / 256, J / 64, B), 256, 0, stream>>>(dT, mx, sei, rs);

  // 7) updates = mean_attn @ v  (WMMA, attn recomputed on the fly)
  wa_attn_v_gemm<<<dim3(D / 128, I / 64, B), 128, 0, stream>>>(dT, mx, sei, rs, vh, out);

  (void)in_sizes; (void)n_in; (void)out_size; (void)ws_size;
}
